// GNNStack_40269613368023
// MI455X (gfx1250) — matrix operand order repaired and checked by reading, not compiled, not run on hardware
//
#include <hip/hip_runtime.h>
#include <hip/hip_bf16.h>
#include <math.h>

typedef __attribute__((ext_vector_type(16))) _Float16 v16h;
typedef __attribute__((ext_vector_type(8)))  _Float16 v8h;
typedef __attribute__((ext_vector_type(8)))  float    v8f;

#define NNODES 50000
#define NEDGES 800000
#define DIN    256
#define HDIM   256
#define LN_EPS 1e-5f

// ---------------------------------------------------------------------------
// utility: grid-stride zero / f32 -> f16 convert
// ---------------------------------------------------------------------------
__global__ void k_zero(float* __restrict__ p, long long n) {
    long long i = (long long)blockIdx.x * blockDim.x + threadIdx.x;
    long long stride = (long long)gridDim.x * blockDim.x;
    for (; i < n; i += stride) p[i] = 0.0f;
}

__global__ void k_cvt16(const float* __restrict__ src, _Float16* __restrict__ dst,
                        long long n) {
    long long i = (long long)blockIdx.x * blockDim.x + threadIdx.x;
    long long stride = (long long)gridDim.x * blockDim.x;
    for (; i < n; i += stride) dst[i] = (_Float16)src[i];
}

// ---------------------------------------------------------------------------
// Pack a row-major f32 weight B[Kd x Nd] into f16 WMMA B-fragment order:
//   frag index = ((ntile*(Kd/32) + kc)*32 + lane)*16 + 2*v + {0,1}
//   maps to    B[(kc*32 + 2*v + 8*(v>>2) + 8*(lane>>4)) * Nd + ntile*16 + (lane&15)]
// ---------------------------------------------------------------------------
__global__ void k_pack_b(const float* __restrict__ B, _Float16* __restrict__ Bp,
                         int Kd, int Nd) {
    int tid = blockIdx.x * blockDim.x + threadIdx.x;
    int kcN = Kd >> 5;
    int total = (Nd >> 4) * kcN * 32 * 8;
    if (tid >= total) return;
    int v    = tid & 7;
    int lane = (tid >> 3) & 31;
    int kc   = (tid >> 8) % kcN;
    int nt   = (tid >> 8) / kcN;
    int k = kc * 32 + 2 * v + ((v >> 2) << 3) + ((lane >> 4) << 3);
    int n = nt * 16 + (lane & 15);
    size_t o = ((size_t)(nt * kcN + kc) * 32 + lane) * 16 + 2 * v;
    Bp[o]     = (_Float16)B[(size_t)k * Nd + n];
    Bp[o + 1] = (_Float16)B[(size_t)(k + 1) * Nd + n];
}

// ---------------------------------------------------------------------------
// degree: deg[col[e]] += ew[e];  dis = rsqrt(deg+1)
// ---------------------------------------------------------------------------
__global__ void k_deg(const long long* __restrict__ ei,
                      const float* __restrict__ ew,
                      float* __restrict__ deg) {
    int e = blockIdx.x * blockDim.x + threadIdx.x;
    if (e >= NEDGES) return;
    long long c = ei[NEDGES + e];
    atomicAdd(&deg[c], ew[e]);
}

__global__ void k_dis(const float* __restrict__ deg, float* __restrict__ dis) {
    int i = blockIdx.x * blockDim.x + threadIdx.x;
    if (i >= NNODES) return;
    dis[i] = rsqrtf(deg[i] + 1.0f);
}

// ---------------------------------------------------------------------------
// A fragment loader (f16 source): two 16B loads of 8 contiguous halves
// (ISA 16-bit A layout: lane holds K runs [k0+8h, +8) and [k0+16+8h, +8))
// ---------------------------------------------------------------------------
__device__ __forceinline__ v16h load_a_frag16(const _Float16* __restrict__ row,
                                              int k0, int half) {
    v8h lo = *(const v8h*)(row + k0 + (half << 3));
    v8h hi = *(const v8h*)(row + k0 + 16 + (half << 3));
    v16h af;
#pragma unroll
    for (int i = 0; i < 8; ++i) { af[i] = lo[i]; af[8 + i] = hi[i]; }
    return af;
}

// ---------------------------------------------------------------------------
// Node GEMM: C[M x 256] = A[M x 256] * B; one wave per 16(M) x 64(N) strip,
// A fragment reused across 4 WMMAs; A f16 row-major, B packed f16 fragments.
// ---------------------------------------------------------------------------
__global__ void k_gemm_node(const _Float16* __restrict__ A,
                            const _Float16* __restrict__ Bp,
                            float* __restrict__ C, int M) {
    const int Kd = HDIM, Nd = HDIM;
    const int kcN = Kd >> 5;              // 8
    int wave  = (blockIdx.x * blockDim.x + threadIdx.x) >> 5;
    int mtile = wave >> 2;                // 4 strips of 64 cols
    int strip = wave & 3;
    if (mtile >= (M >> 4)) return;

    int lane = threadIdx.x & 31;
    int half = lane >> 4;
    int l16  = lane & 15;

    const _Float16* arow = A + (size_t)(mtile * 16 + l16) * Kd;

    v8f acc0 = {}, acc1 = {}, acc2 = {}, acc3 = {};
#pragma unroll
    for (int kc = 0; kc < kcN; ++kc) {
        v16h af = load_a_frag16(arow, kc << 5, half);
        const v16h* bbase =
            (const v16h*)(Bp + ((size_t)((strip * 4) * kcN + kc) * 32 + lane) * 16);
        v16h b0 = bbase[0];
        v16h b1 = bbase[(size_t)kcN * 32];
        v16h b2 = bbase[(size_t)kcN * 64];
        v16h b3 = bbase[(size_t)kcN * 96];
        acc0 = __builtin_amdgcn_wmma_f32_16x16x32_f16(false, af, false, b0, (short)0, acc0, false, false);
        acc1 = __builtin_amdgcn_wmma_f32_16x16x32_f16(false, af, false, b1, (short)0, acc1, false, false);
        acc2 = __builtin_amdgcn_wmma_f32_16x16x32_f16(false, af, false, b2, (short)0, acc2, false, false);
        acc3 = __builtin_amdgcn_wmma_f32_16x16x32_f16(false, af, false, b3, (short)0, acc3, false, false);
    }

    v8f accs[4] = {acc0, acc1, acc2, acc3};
#pragma unroll
    for (int t = 0; t < 4; ++t) {
        int n = (strip * 4 + t) * 16 + l16;
#pragma unroll
        for (int r = 0; r < 8; ++r) {
            int m = mtile * 16 + r + 8 * half;
            C[(size_t)m * Nd + n] = accs[t][r];
        }
    }
}

// ---------------------------------------------------------------------------
// edge scatter: agg[col] += hw[row] * (dis[row]*ew*dis[col]); one block/edge
// ---------------------------------------------------------------------------
__global__ void k_aggregate(const float* __restrict__ hw,
                            const long long* __restrict__ ei,
                            const float* __restrict__ ew,
                            const float* __restrict__ dis,
                            float* __restrict__ agg) {
    int e = blockIdx.x;
    long long r = ei[e];
    long long c = ei[NEDGES + e];
    float norm = dis[r] * ew[e] * dis[c];
    int f = threadIdx.x;
    atomicAdd(&agg[(size_t)c * HDIM + f], hw[(size_t)r * HDIM + f] * norm);
}

// ---------------------------------------------------------------------------
// fused: h = [LN?] relu(agg + hw*dis^2 + b); f32 math, f16 store
// one block (256 threads) per node
// ---------------------------------------------------------------------------
__global__ void k_post(const float* __restrict__ agg,
                       const float* __restrict__ hw,
                       const float* __restrict__ dis,
                       const float* __restrict__ b,
                       const float* __restrict__ g,
                       const float* __restrict__ be,
                       _Float16* __restrict__ hout,
                       int do_ln) {
    int n = blockIdx.x;
    int f = threadIdx.x;
    float d2 = dis[n] * dis[n];
    float v = agg[(size_t)n * HDIM + f] + hw[(size_t)n * HDIM + f] * d2 + b[f];
    v = fmaxf(v, 0.0f);
    if (do_ln) {
        __shared__ float sdata[HDIM];
        sdata[f] = v;
        __syncthreads();
        for (int s = HDIM / 2; s > 0; s >>= 1) {
            if (f < s) sdata[f] += sdata[f + s];
            __syncthreads();
        }
        float mu = sdata[0] * (1.0f / HDIM);
        __syncthreads();
        float d = v - mu;
        sdata[f] = d * d;
        __syncthreads();
        for (int s = HDIM / 2; s > 0; s >>= 1) {
            if (f < s) sdata[f] += sdata[f + s];
            __syncthreads();
        }
        float var = sdata[0] * (1.0f / HDIM);
        v = d * rsqrtf(var + LN_EPS) * g[f] + be[f];
    }
    hout[(size_t)n * HDIM + f] = (_Float16)v;
}

// ---------------------------------------------------------------------------
// edge MLP: one wave per 16 edges, K = 512 = [h[row] | h[col]] (f16).
// Gathered A loaded ONCE into 16 register-resident fragments, reused across
// all 16 N-tiles; inner loop = one 32B packed-B load + one WMMA.
// Bias+ReLU+dot(Wp2) folded per N-tile; lane-reduce; sigmoid.
// ---------------------------------------------------------------------------
__global__ void k_edge_mlp(const _Float16* __restrict__ h,
                           const long long* __restrict__ ei,
                           const _Float16* __restrict__ Bp,  // packed Wp1
                           const float* __restrict__ bp1,    // 256
                           const float* __restrict__ Wp2,    // 256
                           const float* __restrict__ bp2,    // 1
                           float* __restrict__ out) {
    const int kcN = 16;                   // (2*HDIM)/32
    int wave = (blockIdx.x * blockDim.x + threadIdx.x) >> 5;
    if (wave >= NEDGES / 16) return;
    int lane = threadIdx.x & 31;
    int half = lane >> 4;
    int l16  = lane & 15;

    int eA = wave * 16 + l16;
    long long rI = ei[eA];
    long long cI = ei[NEDGES + eA];
    const _Float16* hr = h + (size_t)rI * HDIM;
    const _Float16* hc = h + (size_t)cI * HDIM;

    // Gather the 16x512 A tile once; chunks 0..7 from h[row], 8..15 from h[col]
    v16h afr[16];
#pragma unroll
    for (int kc = 0; kc < 16; ++kc) {
        const _Float16* src = (kc < 8) ? (hr + (kc << 5)) : (hc + ((kc - 8) << 5));
        afr[kc] = load_a_frag16(src, 0, half);
    }

    float logit[8];
#pragma unroll
    for (int r = 0; r < 8; ++r) logit[r] = 0.0f;

    float bias2 = bp2[0];

    for (int nt = 0; nt < 16; ++nt) {
        v8f acc = {};
        const v16h* bbase =
            (const v16h*)(Bp + ((size_t)(nt * kcN) * 32 + lane) * 16);
#pragma unroll
        for (int kc = 0; kc < 16; ++kc) {
            v16h bf = bbase[(size_t)kc * 32];
            acc = __builtin_amdgcn_wmma_f32_16x16x32_f16(
                false, afr[kc], false, bf, (short)0, acc, false, false);
        }
        int n = nt * 16 + l16;
        float bb = bp1[n];
        float w2 = Wp2[n];
#pragma unroll
        for (int r = 0; r < 8; ++r) {
            float v = fmaxf(acc[r] + bb, 0.0f);
            logit[r] += v * w2;
        }
    }

    // reduce the 16 lanes of each half (C-layout: N = lane&15, M = r + 8*half)
#pragma unroll
    for (int r = 0; r < 8; ++r) {
        float s = logit[r];
        for (int off = 1; off < 16; off <<= 1) s += __shfl_xor(s, off);
        if (l16 == 0) {
            int e = wave * 16 + r + 8 * half;
            float z = s + bias2;
            out[e] = 1.0f / (1.0f + expf(-z));
        }
    }
}

// ---------------------------------------------------------------------------
// host
// ---------------------------------------------------------------------------
extern "C" void kernel_launch(void* const* d_in, const int* in_sizes, int n_in,
                              void* d_out, int out_size, void* d_ws, size_t ws_size,
                              hipStream_t stream) {
    const float*     x   = (const float*)d_in[0];
    const long long* ei  = (const long long*)d_in[1];
    const float*     ea  = (const float*)d_in[2];
    const float*     W[3]  = { (const float*)d_in[3], (const float*)d_in[5], (const float*)d_in[7] };
    const float*     bb[3] = { (const float*)d_in[4], (const float*)d_in[6], (const float*)d_in[8] };
    const float*     g[2]  = { (const float*)d_in[9],  (const float*)d_in[11] };
    const float*     be[2] = { (const float*)d_in[10], (const float*)d_in[12] };
    const float*     Wp1 = (const float*)d_in[13];
    const float*     bp1 = (const float*)d_in[14];
    const float*     Wp2 = (const float*)d_in[15];
    const float*     bp2 = (const float*)d_in[16];
    float* out = (float*)d_out;

    const size_t NH = (size_t)NNODES * HDIM;
    float* ws  = (float*)d_ws;
    float* deg = ws;                        // N f32
    float* dis = deg + NNODES;              // N f32
    float* hw  = dis + NNODES;              // N*H f32
    float* agg = hw  + NH;                  // N*H f32
    _Float16* h  = (_Float16*)(agg + NH);   // N*H f16
    _Float16* xh = h + NH;                  // N*H f16
    _Float16* wpk[3];
    wpk[0] = xh + NH;
    wpk[1] = wpk[0] + (size_t)HDIM * HDIM;
    wpk[2] = wpk[1] + (size_t)HDIM * HDIM;
    _Float16* wp1pk = wpk[2] + (size_t)HDIM * HDIM;   // 2*HDIM*HDIM halves

    // pack weights into f16 WMMA fragment order; convert x -> f16
    for (int l = 0; l < 3; ++l) {
        int total = (HDIM / 16) * (HDIM / 32) * 32 * 8;      // 32768
        k_pack_b<<<(total + 255) / 256, 256, 0, stream>>>(W[l], wpk[l], HDIM, HDIM);
    }
    {
        int total = (HDIM / 16) * (2 * HDIM / 32) * 32 * 8;  // 65536
        k_pack_b<<<(total + 255) / 256, 256, 0, stream>>>(Wp1, wp1pk, 2 * HDIM, HDIM);
    }
    k_cvt16<<<4096, 256, 0, stream>>>(x, xh, (long long)NH);

    // degree + symmetric norm
    k_zero<<<1024, 256, 0, stream>>>(deg, NNODES);
    k_deg<<<(NEDGES + 255) / 256, 256, 0, stream>>>(ei, ea, deg);
    k_dis<<<(NNODES + 255) / 256, 256, 0, stream>>>(deg, dis);

    const int gemm_waves  = (NNODES / 16) * 4;               // 12500 strips
    const int gemm_blocks = (gemm_waves * 32 + 255) / 256;

    for (int l = 0; l < 3; ++l) {
        const _Float16* hin = (l == 0) ? xh : h;
        k_zero<<<4096, 256, 0, stream>>>(agg, (long long)NH);
        k_gemm_node<<<gemm_blocks, 256, 0, stream>>>(hin, wpk[l], hw, NNODES);
        k_aggregate<<<NEDGES, HDIM, 0, stream>>>(hw, ei, ea, dis, agg);
        int do_ln = (l < 2) ? 1 : 0;
        const float* gl  = do_ln ? g[l]  : g[0];   // unused when !do_ln
        const float* bel = do_ln ? be[l] : be[0];
        k_post<<<NNODES, HDIM, 0, stream>>>(agg, hw, dis, bb[l], gl, bel, h, do_ln);
    }

    const int mlp_waves  = NEDGES / 16;                      // 50000
    const int mlp_blocks = (mlp_waves * 32 + 255) / 256;
    k_edge_mlp<<<mlp_blocks, 256, 0, stream>>>(h, ei, wp1pk, bp1, Wp2, bp2, out);
}
